// SchNet__41016937677233
// MI455X (gfx1250) — compile-verified, matrix operand order checked by reading, not run on hardware
//
#include <hip/hip_runtime.h>
#include <hip/hip_bf16.h>
#include <math.h>

// ---------------------------------------------------------------------------
// SchNet interaction stack for MI455X (gfx1250), wave32 + WMMA f16.
// All WMMA operand tiles are kept in LDS pre-swizzled into the per-lane
// register layout (ISA 7.12.2), so fragment loads are contiguous b128 reads.
// ---------------------------------------------------------------------------

#define NNODES   10000
#define NEDGES   320000
#define HID      128          // HIDDEN == N_FILTERS == 128
#define NG       50
#define NGPAD    64           // gaussians padded to 64 for K%32==0
#define NBLK     6
#define CUTOFF   10.0f
#define LOG2F_   0.69314718055994530942f
#define LOG2E_   1.44269504088896340736f
#define PI_      3.14159265358979323846f

typedef __attribute__((ext_vector_type(16))) _Float16 v16h;
typedef __attribute__((ext_vector_type(8)))  float    v8f;

union HalfPair { _Float16 h[2]; unsigned u; };

// raw hardware base-2 transcendentals (no denorm guard sequences)
#if defined(__has_builtin)
#  if __has_builtin(__builtin_amdgcn_exp2f)
#    define EXP2F(x) __builtin_amdgcn_exp2f(x)
#  endif
#  if __has_builtin(__builtin_amdgcn_logf)
#    define LOG2F(x) __builtin_amdgcn_logf(x)
#  endif
#endif
#ifndef EXP2F
#  define EXP2F(x) exp2f(x)
#endif
#ifndef LOG2F
#  define LOG2F(x) log2f(x)
#endif

// shifted softplus: softplus(x) - ln2, overflow-safe.
// log2 argument is 1 + 2^(-|x|*log2e) in (1,2] -> never denormal, raw v_log ok.
__device__ __forceinline__ float ssp(float x) {
    float t = EXP2F(-fabsf(x) * LOG2E_);          // e^{-|x|}
    return fmaf(LOG2F(1.0f + t) - 1.0f, LOG2F_, fmaxf(x, 0.0f));
}

// Offset (elements) of (r, k) inside one 16x32 f16 WMMA fragment block,
// stored so lane L holds its 16 values contiguously at [L*16 .. L*16+15].
// r = M (A operand) or N-local (B operand); k in [0,32).
// ISA 7.12.2: lane = r + 16*(k&8 ? 1:0); pair idx = (k&16 ? 8:0) + (k&7).
__device__ __forceinline__ int frag_off(int r, int k) {
    return ((r + ((k & 8) << 1)) << 4) + (((k & 16) >> 1) | (k & 7));
}

#define WMMA_F16(a, b, c) \
    __builtin_amdgcn_wmma_f32_16x16x32_f16(false, (a), false, (b), (short)0, (c), false, false)

// ---------------------------------------------------------------------------
// prep: per-edge distance + cosine cutoff envelope
// ---------------------------------------------------------------------------
__global__ __launch_bounds__(256) void prep_kernel(
    const float* __restrict__ pos, const int* __restrict__ eidx,
    float* __restrict__ dE, float* __restrict__ cE)
{
    int e = blockIdx.x * 256 + threadIdx.x;
    if (e >= NEDGES) return;
    int r = eidx[e], c = eidx[NEDGES + e];
    float dx = pos[r * 3 + 0] - pos[c * 3 + 0];
    float dy = pos[r * 3 + 1] - pos[c * 3 + 1];
    float dz = pos[r * 3 + 2] - pos[c * 3 + 2];
    float d  = sqrtf(dx * dx + dy * dy + dz * dz);
    dE[e] = d;
    cE[e] = 0.5f * (cosf(d * PI_ / CUTOFF) + 1.0f);
}

// ---------------------------------------------------------------------------
// weight conversion f32 -> f16, written directly in WMMA-fragment order:
// dst layout per block: [(K/32) k-steps][8 n-tiles][512] f16.
// Also pads K (mlp_w1: 50 -> 64).
// ---------------------------------------------------------------------------
__global__ __launch_bounds__(256) void swz_w_kernel(
    const float* __restrict__ src, _Float16* __restrict__ dst, int Kpad, int Ksrc)
{
    int i = blockIdx.x * 256 + threadIdx.x;
    const int per = Kpad * HID;
    if (i >= NBLK * per) return;
    int b = i / per, rem = i - b * per;
    int k = rem >> 7, n = rem & 127;
    float v = (k < Ksrc) ? src[((size_t)b * Ksrc + k) * HID + n] : 0.0f;
    dst[(size_t)b * per + (((k >> 5) << 3) + (n >> 4)) * 512 + frag_off(n & 15, k & 31)]
        = (_Float16)v;
}

__global__ __launch_bounds__(256) void copy_f32_kernel(
    const float* __restrict__ src, float* __restrict__ dst, int n)
{
    int i = blockIdx.x * 256 + threadIdx.x;
    if (i < n) dst[i] = src[i];
}

__global__ __launch_bounds__(256) void zero_kernel(float* __restrict__ p, int n)
{
    int i = blockIdx.x * 256 + threadIdx.x;
    if (i < n) p[i] = 0.0f;
}

// ---------------------------------------------------------------------------
// generic node GEMM: out[16-row tiles] = A[N x 128] @ W[128 x 128]
//   MODE 0: outF = A@W                        (xf = h @ lin1_w)
//   MODE 1: Osw  = f16(ssp(A@W + bias))       (t = ssp(agg @ lin2_w + b)),
//            written pre-swizzled per 16-row tile for MODE 2's A staging
//   MODE 2: resid += A@W + bias               (h += t @ lin_w + b), A = Osw
// 256 threads = 8 waves; wave w covers output columns [16w, 16w+16).
// ---------------------------------------------------------------------------
template <int MODE>
__global__ __launch_bounds__(256) void node_gemm_kernel(
    const float* __restrict__ Af, const _Float16* __restrict__ Asw,
    const _Float16* __restrict__ Wsw, const float* __restrict__ bias,
    float* __restrict__ outF, _Float16* __restrict__ Osw,
    float* __restrict__ resid, int nRows)
{
    __shared__ __align__(32) _Float16 sW[HID * HID];   // 32 KB, fragment order
    __shared__ __align__(32) _Float16 sA[16 * HID];    //  4 KB, fragment order

    const int tid  = threadIdx.x;
    const int lane = tid & 31;
    const int wv   = tid >> 5;

    {   // stage pre-swizzled weights: contiguous 128-bit copies
        const uint4* src = (const uint4*)Wsw;
        uint4* dst = (uint4*)sW;
#pragma unroll 4
        for (int i = tid; i < (HID * HID) / 8; i += 256) dst[i] = src[i];
    }
    __syncthreads();

    const int nTiles = nRows >> 4;   // nRows multiple of 16
    for (int tile = blockIdx.x; tile < nTiles; tile += gridDim.x) {
        const int rowBase = tile << 4;
        if constexpr (MODE == 2) {
            const uint4* src = (const uint4*)(Asw + (size_t)tile * (16 * HID));
            uint4* dst = (uint4*)sA;
            for (int i = tid; i < (16 * HID) / 8; i += 256) dst[i] = src[i];
        } else {
            // f32 row-major -> f16 fragment order (pairs -> one b32 store)
            for (int idx = tid; idx < (16 * HID) / 2; idx += 256) {
                const int r  = idx >> 6;
                const int kp = (idx & 63) << 1;
                HalfPair hp;
                hp.h[0] = (_Float16)Af[(rowBase + r) * HID + kp];
                hp.h[1] = (_Float16)Af[(rowBase + r) * HID + kp + 1];
                *(unsigned*)&sA[(kp >> 5) * 512 + frag_off(r, kp & 31)] = hp.u;
            }
        }
        __syncthreads();

        v8f acc = {};
#pragma unroll
        for (int kk = 0; kk < 4; ++kk) {
            v16h a = *(const v16h*)(sA + kk * 512 + lane * 16);
            v16h b = *(const v16h*)(sW + ((kk << 3) + wv) * 512 + lane * 16);
            acc = WMMA_F16(a, b, acc);
        }

        const int n = (wv << 4) + (lane & 15);
#pragma unroll
        for (int r = 0; r < 8; ++r) {
            const int m = r + ((lane >> 4) << 3);
            float v = acc[r];
            if constexpr (MODE == 0) {
                outF[(rowBase + m) * HID + n] = v;
            } else if constexpr (MODE == 1) {
                Osw[(size_t)tile * (16 * HID) + (n >> 5) * 512 + frag_off(m, n & 31)]
                    = (_Float16)ssp(v + bias[n]);
            } else {
                resid[(rowBase + m) * HID + n] += v + bias[n];
            }
        }
        __syncthreads();
    }
}

// ---------------------------------------------------------------------------
// fused edge pipeline: gaussian smearing -> MLP1 -> ssp -> MLP2 -> cutoff
//                      -> gather xf[row] -> atomic scatter-add into agg[col]
// Tile = 16 edges; 8 waves cover the 128 filter columns. Weights resident in
// LDS (fragment order), persistent loop over edge tiles.
// ---------------------------------------------------------------------------
__global__ __launch_bounds__(256) void edge_kernel(
    const int* __restrict__ eRow, const int* __restrict__ eCol,
    const float* __restrict__ dE, const float* __restrict__ cE,
    const _Float16* __restrict__ W1, const _Float16* __restrict__ W2,
    const float* __restrict__ b1, const float* __restrict__ b2,
    const float* __restrict__ xf, float* __restrict__ agg)
{
    __shared__ __align__(32) _Float16 sW1[NGPAD * HID];   // 16 KB, fragment order
    __shared__ __align__(32) _Float16 sW2[HID * HID];     // 32 KB, fragment order
    __shared__ __align__(32) _Float16 sAttr[16 * NGPAD];  //  2 KB, fragment order
    __shared__ __align__(32) _Float16 sH1[16 * HID];      //  4 KB, fragment order
    __shared__ int   sRow[16];
    __shared__ int   sDst[16];
    __shared__ float sCc[16];

    const int tid  = threadIdx.x;
    const int lane = tid & 31;
    const int wv   = tid >> 5;

    const float gstep  = CUTOFF / (float)(NG - 1);
    const float coeff2 = (-0.5f / (gstep * gstep)) * LOG2E_;  // exp -> exp2

    {   // stage both weight matrices: contiguous 128-bit copies
        const uint4* s1 = (const uint4*)W1;
        uint4* d1 = (uint4*)sW1;
#pragma unroll 4
        for (int i = tid; i < (NGPAD * HID) / 8; i += 256) d1[i] = s1[i];
        const uint4* s2 = (const uint4*)W2;
        uint4* d2 = (uint4*)sW2;
#pragma unroll 4
        for (int i = tid; i < (HID * HID) / 8; i += 256) d2[i] = s2[i];
    }
    __syncthreads();

    const int nTiles = NEDGES >> 4;
    for (int tile = blockIdx.x; tile < nTiles; tile += gridDim.x) {
        const int eBase = tile << 4;

        if (tid < 16) {
            const int e = eBase + tid;
            sRow[tid] = eRow[e];
            sDst[tid] = eCol[e];
            sCc[tid]  = cE[e];
            __builtin_prefetch(&xf[sRow[tid] * HID], 0, 3);  // global_prefetch_b8
        }
        // gaussian smearing straight into fragment order (pairs -> b32 store)
        for (int idx = tid; idx < (16 * NGPAD) / 2; idx += 256) {
            const int el = idx >> 5;
            const int gp = (idx & 31) << 1;
            const float d = dE[eBase + el];
            HalfPair hp;
#pragma unroll
            for (int j = 0; j < 2; ++j) {
                const int g = gp + j;
                float v = 0.0f;
                if (g < NG) {
                    const float t = d - (float)g * gstep;
                    v = EXP2F(coeff2 * t * t);
                }
                hp.h[j] = (_Float16)v;
            }
            *(unsigned*)&sAttr[(gp >> 5) * 512 + frag_off(el, gp & 31)] = hp.u;
        }
        __syncthreads();

        // ---- GEMM1: [16 x 64] @ [64 x 128] + b1, ssp -> sH1 (fragment order)
        {
            v8f acc = {};
#pragma unroll
            for (int kk = 0; kk < 2; ++kk) {
                v16h a = *(const v16h*)(sAttr + kk * 512 + lane * 16);
                v16h b = *(const v16h*)(sW1 + ((kk << 3) + wv) * 512 + lane * 16);
                acc = WMMA_F16(a, b, acc);
            }
            const int n = (wv << 4) + (lane & 15);
#pragma unroll
            for (int r = 0; r < 8; ++r) {
                const int m = r + ((lane >> 4) << 3);
                sH1[(n >> 5) * 512 + frag_off(m, n & 31)] = (_Float16)ssp(acc[r] + b1[n]);
            }
        }
        __syncthreads();

        // ---- GEMM2: [16 x 128] @ [128 x 128] + b2, * C, * xf[row], scatter
        {
            v8f acc = {};
#pragma unroll
            for (int kk = 0; kk < 4; ++kk) {
                v16h a = *(const v16h*)(sH1 + kk * 512 + lane * 16);
                v16h b = *(const v16h*)(sW2 + ((kk << 3) + wv) * 512 + lane * 16);
                acc = WMMA_F16(a, b, acc);
            }
            const int n = (wv << 4) + (lane & 15);
#pragma unroll
            for (int r = 0; r < 8; ++r) {
                const int m = r + ((lane >> 4) << 3);
                float w = (acc[r] + b2[n]) * sCc[m];
                float msg = w * xf[sRow[m] * HID + n];
                atomicAdd(&agg[sDst[m] * HID + n], msg);  // segment_sum
            }
        }
        __syncthreads();
    }
}

// ---------------------------------------------------------------------------
// finalize: d_out = concat(h [N*H f32], edge_index [2*E i32 bits])
// ---------------------------------------------------------------------------
__global__ __launch_bounds__(256) void finalize_kernel(
    const float* __restrict__ h, const int* __restrict__ eidx, void* __restrict__ out)
{
    int i = blockIdx.x * 256 + threadIdx.x;
    const int nh = NNODES * HID;
    if (i < nh) {
        ((float*)out)[i] = h[i];
    } else if (i < nh + 2 * NEDGES) {
        ((int*)out)[i] = eidx[i - nh];
    }
}

// ---------------------------------------------------------------------------
extern "C" void kernel_launch(void* const* d_in, const int* in_sizes, int n_in,
                              void* d_out, int out_size, void* d_ws, size_t ws_size,
                              hipStream_t stream)
{
    const float* h_in    = (const float*)d_in[0];
    const float* pos     = (const float*)d_in[1];
    const int*   eidx    = (const int*)d_in[2];
    const float* mlp_w1  = (const float*)d_in[3];
    const float* mlp_b1  = (const float*)d_in[4];
    const float* mlp_w2  = (const float*)d_in[5];
    const float* mlp_b2  = (const float*)d_in[6];
    const float* lin1_w  = (const float*)d_in[7];
    const float* lin2_w  = (const float*)d_in[8];
    const float* lin2_b  = (const float*)d_in[9];
    const float* lin_w   = (const float*)d_in[10];
    const float* lin_b   = (const float*)d_in[11];

    char* ws = (char*)d_ws;
    size_t off = 0;
    auto carve = [&](size_t bytes) -> char* {
        char* p = ws + off;
        off = (off + bytes + 255) & ~(size_t)255;
        return p;
    };
    const int NH = NNODES * HID;
    float*    h_cur = (float*)carve((size_t)NH * 4);
    float*    xf    = (float*)carve((size_t)NH * 4);
    float*    agg   = (float*)carve((size_t)NH * 4);
    _Float16* th    = (_Float16*)carve((size_t)NH * 2);   // swizzled per-tile
    float*    dE    = (float*)carve((size_t)NEDGES * 4);
    float*    cEnv  = (float*)carve((size_t)NEDGES * 4);
    _Float16* w1h   = (_Float16*)carve((size_t)NBLK * NGPAD * HID * 2);
    _Float16* w2h   = (_Float16*)carve((size_t)NBLK * HID * HID * 2);
    _Float16* l1h   = (_Float16*)carve((size_t)NBLK * HID * HID * 2);
    _Float16* l2h   = (_Float16*)carve((size_t)NBLK * HID * HID * 2);
    _Float16* lnh   = (_Float16*)carve((size_t)NBLK * HID * HID * 2);

    // one-time prep
    prep_kernel<<<(NEDGES + 255) / 256, 256, 0, stream>>>(pos, eidx, dE, cEnv);
    swz_w_kernel<<<(NBLK * NGPAD * HID + 255) / 256, 256, 0, stream>>>(mlp_w1, w1h, NGPAD, NG);
    const int WN = NBLK * HID * HID;
    swz_w_kernel<<<(WN + 255) / 256, 256, 0, stream>>>(mlp_w2, w2h, HID, HID);
    swz_w_kernel<<<(WN + 255) / 256, 256, 0, stream>>>(lin1_w, l1h, HID, HID);
    swz_w_kernel<<<(WN + 255) / 256, 256, 0, stream>>>(lin2_w, l2h, HID, HID);
    swz_w_kernel<<<(WN + 255) / 256, 256, 0, stream>>>(lin_w, lnh, HID, HID);
    copy_f32_kernel<<<(NH + 255) / 256, 256, 0, stream>>>(h_in, h_cur, NH);

    const int nodeTiles = NNODES / 16;  // 625
    for (int b = 0; b < NBLK; ++b) {
        // xf = h @ lin1_w[b]
        node_gemm_kernel<0><<<nodeTiles, 256, 0, stream>>>(
            h_cur, nullptr, l1h + (size_t)b * HID * HID, nullptr,
            xf, nullptr, nullptr, NNODES);
        // agg = 0
        zero_kernel<<<(NH + 255) / 256, 256, 0, stream>>>(agg, NH);
        // fused edge pipeline + scatter-add
        edge_kernel<<<2000, 256, 0, stream>>>(
            eidx, eidx + NEDGES, dE, cEnv,
            w1h + (size_t)b * NGPAD * HID, w2h + (size_t)b * HID * HID,
            mlp_b1 + (size_t)b * HID, mlp_b2 + (size_t)b * HID,
            xf, agg);
        // t = ssp(agg @ lin2_w[b] + lin2_b[b])   (f16, fragment order)
        node_gemm_kernel<1><<<nodeTiles, 256, 0, stream>>>(
            agg, nullptr, l2h + (size_t)b * HID * HID, lin2_b + (size_t)b * HID,
            nullptr, th, nullptr, NNODES);
        // h += t @ lin_w[b] + lin_b[b]
        node_gemm_kernel<2><<<nodeTiles, 256, 0, stream>>>(
            nullptr, th, lnh + (size_t)b * HID * HID, lin_b + (size_t)b * HID,
            nullptr, nullptr, h_cur, NNODES);
    }

    finalize_kernel<<<(NH + 2 * NEDGES + 255) / 256, 256, 0, stream>>>(h_cur, eidx, d_out);
}
